// VanillaLstmEncDec_37271726195511
// MI455X (gfx1250) — matrix operand order repaired and checked by reading, not compiled, not run on hardware
//
#include <hip/hip_runtime.h>
#include <math.h>

typedef __attribute__((ext_vector_type(16))) _Float16 v16h;
typedef __attribute__((ext_vector_type(8)))  _Float16 v8h;
typedef __attribute__((ext_vector_type(2)))  __fp16   v2hf;  /* matches cvt_pkrtz */
typedef __attribute__((ext_vector_type(8)))  float    v8f;

#define B_TOT    131072
#define OBS_LEN  8
#define PRED_LEN 12
#define EMB      128
#define HID      128
#define ZCOLS    512      /* 4*HID */

#define WAVES_PER_BLOCK 4
#define BLOCK_THREADS   (WAVES_PER_BLOCK * 32)
#define ROWS_PER_WAVE   16
#define ROWS_PER_BLOCK  (WAVES_PER_BLOCK * ROWS_PER_WAVE)   /* 64  */
#define NBLOCKS         (B_TOT / ROWS_PER_BLOCK)            /* 2048 */

/* ---------------- LDS layout (bytes) ---------------- */
#define WT_STRIDE      528                           /* 256 f16 + 16B pad       */
#define ZB_OFF         (512 * WT_STRIDE)             /* f32[512] gate bias      */
#define EW_OFF         (ZB_OFF + 512 * 4)            /* f32[256] embed W (2x128)*/
#define EB_OFF         (EW_OFF + 256 * 4)            /* f32[128] embed bias     */
#define WO_OFF         (EB_OFF + 128 * 4)            /* f32[256] hidden2pos     */
#define BO_OFF         (WO_OFF + 256 * 4)            /* f32[2]                  */
#define STAGE_OFF      (BO_OFF + 16)
#define STAGE_STRIDE   784                           /* (x|h0|h1) f16 + pad     */
#define STAGE_PER_WAVE (16 * STAGE_STRIDE)
#define SMEM_BYTES     (STAGE_OFF + WAVES_PER_BLOCK * STAGE_PER_WAVE) /* 325136 */

/* fast sigmoid/tanh: v_exp_f32 + v_rcp_f32 (both TRANS, co-execute with WMMA) */
__device__ __forceinline__ float fast_sig(float x) {
    return __builtin_amdgcn_rcpf(1.0f + __expf(-x));
}
__device__ __forceinline__ float fast_tanh(float x) {
    float e = __expf(2.0f * x);
    return (e - 1.0f) * __builtin_amdgcn_rcpf(e + 1.0f);
}

__device__ __forceinline__ v8f splat8(float x) {
    v8f r;
#pragma unroll
    for (int i = 0; i < 8; ++i) r[i] = x;
    return r;
}

/* Build a 16xf16 WMMA operand from two 16-byte LDS chunks. */
__device__ __forceinline__ v16h load2x8h(const char* p0, const char* p1) {
    v8h lo = *(const v8h*)p0;
    v8h hi = *(const v8h*)p1;
    v16h r;
#pragma unroll
    for (int i = 0; i < 8; ++i) { r[i] = lo[i]; r[i + 8] = hi[i]; }
    return r;
}

__global__ void __launch_bounds__(BLOCK_THREADS, 1)
lstm_encdec_kernel(const float* __restrict__ obs,
                   const float* __restrict__ We,    const float* __restrict__ be,
                   const float* __restrict__ Wih_e, const float* __restrict__ Whh_e,
                   const float* __restrict__ b_e,
                   const float* __restrict__ Wd,    const float* __restrict__ bd,
                   const float* __restrict__ Wih_d, const float* __restrict__ Whh_d,
                   const float* __restrict__ b_d,
                   const float* __restrict__ Wo,    const float* __restrict__ bo,
                   float* __restrict__ out)
{
    extern __shared__ char smem[];

    const int tid  = threadIdx.x;
    const int wave = tid >> 5;
    const int lane = tid & 31;
    const int lrow = lane & 15;   /* tile row / N-col within 16 */
    const int lhi  = lane >> 4;   /* half-wave select           */
    const int row_base = blockIdx.x * ROWS_PER_BLOCK + wave * ROWS_PER_WAVE;

    char*  stage = smem + STAGE_OFF + wave * STAGE_PER_WAVE;
    float* zbias = (float*)(smem + ZB_OFF);
    float* eW    = (float*)(smem + EW_OFF);
    float* eB    = (float*)(smem + EB_OFF);
    float* woL   = (float*)(smem + WO_OFF);
    float* boL   = (float*)(smem + BO_OFF);

    /* ---- cooperative load of one phase's weights into LDS (W^T as f16) ---- */
    auto load_phase = [&](const float* Wih, const float* Whh, const float* bz,
                          const float* Wemb, const float* bemb) {
        for (int idx = tid; idx < 512 * 128; idx += BLOCK_THREADS) {
            int n  = idx & 511;           /* gate column            */
            int k  = (idx >> 9) * 2;      /* concat K row (pair)    */
            float v0 = (k     < EMB) ? Wih[k * ZCOLS + n]       : Whh[(k - EMB) * ZCOLS + n];
            float v1 = (k + 1 < EMB) ? Wih[(k + 1) * ZCOLS + n] : Whh[(k + 1 - EMB) * ZCOLS + n];
            v2hf p = __builtin_amdgcn_cvt_pkrtz(v0, v1);
            *((v2hf*)(smem + n * WT_STRIDE + k * 2)) = p;
        }
        for (int i = tid; i < ZCOLS; i += BLOCK_THREADS) zbias[i] = bz[i];
        for (int i = tid; i < 256;   i += BLOCK_THREADS) eW[i]    = Wemb[i];
        for (int i = tid; i < EMB;   i += BLOCK_THREADS) eB[i]    = bemb[i];
    };

    for (int i = tid; i < 256; i += BLOCK_THREADS) woL[i] = Wo[i];
    if (tid < 2) boL[tid] = bo[tid];

    load_phase(Wih_e, Whh_e, b_e, We, be);
    __syncthreads();

    /* ---- per-wave state: c in registers (C-layout), h in LDS (double buffered) ---- */
    v8f cst[8];
#pragma unroll
    for (int t = 0; t < 8; ++t) cst[t] = splat8(0.0f);

    int cur = 0;
    {   /* zero h buffer 0 */
        char* hrow = stage + lrow * STAGE_STRIDE + (128 + lhi * 64) * 2;
        v8h z;
#pragma unroll
        for (int i = 0; i < 8; ++i) z[i] = (_Float16)0.0f;
#pragma unroll
        for (int i = 0; i < 8; ++i) *((v8h*)(hrow + i * 16)) = z;
    }

    /* ---- embed (px,py) -> x region of stage (f16, packed stores) ---- */
    auto embed = [&](float px, float py) {
        char* xrow = stage + lrow * STAGE_STRIDE;
        int c0 = lhi * 64;
#pragma unroll
        for (int e = 0; e < 64; e += 2) {
            int e0 = c0 + e;
            float v0 = px * eW[e0]     + py * eW[128 + e0]     + eB[e0];
            float v1 = px * eW[e0 + 1] + py * eW[128 + e0 + 1] + eB[e0 + 1];
            *((v2hf*)(xrow + e0 * 2)) = __builtin_amdgcn_cvt_pkrtz(v0, v1);
        }
    };

    /* ---- one LSTM step: z = [x|h]@W^T (+b in epilogue), gates, h -> other buffer ---- */
    auto lstm_step = [&](int curbuf) -> int {
        /* A slices: 8 x (16x32 f16); slices 0-3 from x, 4-7 from h[curbuf] */
        v16h a[8];
#pragma unroll
        for (int ks = 0; ks < 8; ++ks) {
            int colbase = (ks < 4) ? (ks * 32) : (128 + curbuf * 128 + (ks - 4) * 32);
            const char* base = stage + lrow * STAGE_STRIDE + (colbase + lhi * 8) * 2;
            a[ks] = load2x8h(base, base + 32);
        }
        const int nxt = curbuf ^ 1;
#pragma unroll 1
        for (int t = 0; t < 8; ++t) {
            /* zero C folds to WMMA inline SRC2=0; bias added in epilogue */
            v8f acc[4] = {};
#pragma unroll
            for (int ks = 0; ks < 8; ++ks) {
                /* batch the four gate B-tiles into distinct buffers, then issue
                 * four independent WMMAs back-to-back (separates load/consume
                 * register lifetimes -> fewer WAR hazard nops). */
                v16h bt[4];
#pragma unroll
                for (int g = 0; g < 4; ++g) {
                    int n = g * HID + t * 16 + lrow;   /* W^T row for this lane */
                    const char* bb = smem + n * WT_STRIDE + (ks * 32 + lhi * 8) * 2;
                    bt[g] = load2x8h(bb, bb + 32);
                }
#pragma unroll
                for (int g = 0; g < 4; ++g)
                    acc[g] = __builtin_amdgcn_wmma_f32_16x16x32_f16(
                        false, a[ks], false, bt[g], (short)0, acc[g], false, false);
            }
            /* per-lane gate biases (one scalar LDS read per gate) */
            float bi  = zbias[0 * HID + t * 16 + lrow];
            float bf  = zbias[1 * HID + t * 16 + lrow];
            float bg  = zbias[2 * HID + t * 16 + lrow];
            float bo_ = zbias[3 * HID + t * 16 + lrow];
            /* gate nonlinearities on the C-layout accumulators */
#pragma unroll
            for (int v = 0; v < 8; ++v) {
                float iv = fast_sig(acc[0][v] + bi);
                float fv = fast_sig(acc[1][v] + bf);
                float gv = fast_tanh(acc[2][v] + bg);
                float ov = fast_sig(acc[3][v] + bo_);
                float cc = fv * cst[t][v] + iv * gv;
                cst[t][v] = cc;
                float hh = ov * fast_tanh(cc);
                /* C layout: VGPR v -> row v + lhi*8, col lrow (within tile t) */
                *((_Float16*)(stage + (v + lhi * 8) * STAGE_STRIDE
                              + (128 + nxt * 128 + t * 16 + lrow) * 2)) = (_Float16)hh;
            }
        }
        return nxt;
    };

    /* ---- hidden2pos: lane (m=lrow, d=lhi) -> pos[m,d] ---- */
    auto project = [&](int curbuf) -> float {
        const char* hrow = stage + lrow * STAGE_STRIDE + (128 + curbuf * 128) * 2;
        float acc = boL[lhi];
#pragma unroll
        for (int k8 = 0; k8 < 16; ++k8) {
            v8h hv = *((const v8h*)(hrow + k8 * 16));
#pragma unroll
            for (int j = 0; j < 8; ++j)
                acc += (float)hv[j] * woL[(k8 * 8 + j) * 2 + lhi];
        }
        return acc;
    };

    /* ================= encoder ================= */
#pragma unroll 1
    for (int t = 0; t < OBS_LEN; ++t) {
        const float* op = obs + ((size_t)t * B_TOT + row_base + lrow) * 2;
        float px = op[0], py = op[1];
        if (t + 1 < OBS_LEN)
            __builtin_prefetch(obs + ((size_t)(t + 1) * B_TOT + row_base + lrow) * 2, 0, 0);
        embed(px, py);
        cur = lstm_step(cur);
    }

    float pos = project(cur);                 /* pos0 from h_enc */
    out[((size_t)0 * B_TOT + row_base + lrow) * 2 + lhi] = pos;

    /* ================= decoder ================= */
    __syncthreads();                          /* all waves done with encoder weights */
    load_phase(Wih_d, Whh_d, b_d, Wd, bd);
    __syncthreads();

#pragma unroll
    for (int t = 0; t < 8; ++t) cst[t] = splat8(0.0f);   /* c_d = 0, h_d = h_enc */

#pragma unroll 1
    for (int s = 1; s < PRED_LEN; ++s) {
        float px = __shfl(pos, lrow, 32);        /* pos[m,0] lives in lane m    */
        float py = __shfl(pos, lrow + 16, 32);   /* pos[m,1] lives in lane m+16 */
        embed(px, py);
        cur = lstm_step(cur);
        pos = project(cur);
        out[((size_t)s * B_TOT + row_base + lrow) * 2 + lhi] = pos;
    }
}

extern "C" void kernel_launch(void* const* d_in, const int* in_sizes, int n_in,
                              void* d_out, int out_size, void* d_ws, size_t ws_size,
                              hipStream_t stream) {
    (void)in_sizes; (void)n_in; (void)d_ws; (void)ws_size; (void)out_size;

    const float* obs   = (const float*)d_in[0];
    const float* We    = (const float*)d_in[1];
    const float* be    = (const float*)d_in[2];
    const float* Wih_e = (const float*)d_in[3];
    const float* Whh_e = (const float*)d_in[4];
    const float* b_e   = (const float*)d_in[5];
    const float* Wd    = (const float*)d_in[6];
    const float* bd    = (const float*)d_in[7];
    const float* Wih_d = (const float*)d_in[8];
    const float* Whh_d = (const float*)d_in[9];
    const float* b_d   = (const float*)d_in[10];
    const float* Wo    = (const float*)d_in[11];
    const float* bo    = (const float*)d_in[12];
    float* out = (float*)d_out;

    (void)hipFuncSetAttribute((const void*)lstm_encdec_kernel,
                              hipFuncAttributeMaxDynamicSharedMemorySize, SMEM_BYTES);

    lstm_encdec_kernel<<<NBLOCKS, BLOCK_THREADS, SMEM_BYTES, stream>>>(
        obs, We, be, Wih_e, Whh_e, b_e, Wd, bd, Wih_d, Whh_d, b_d, Wo, bo, out);
}
